// MultiheadAttention_44530220925681
// MI455X (gfx1250) — compile-verified, matrix operand order checked by reading, not run on hardware
//
#include <hip/hip_runtime.h>

// ---------------------------------------------------------------------------
// MultiheadAttention forward for MI455X (gfx1250), wave32 + WMMA bf16.
// B=4, S=2048, D=1024, H=16, DH=64.
//   - fp32 -> bf16 conversion pass
//   - QKV projections + output projection: register-double-buffered WMMA GEMM
//   - attention: flash-style, K/V tiles staged in LDS with
//     global_load_async_to_lds_b128 (ASYNCcnt) double buffering
// ---------------------------------------------------------------------------

typedef __attribute__((ext_vector_type(16))) __bf16 v16bf;
typedef __attribute__((ext_vector_type(8)))  float  v8f;
typedef unsigned short u16;

#define BB 4
#define SS 2048
#define DD 1024
#define HH 16
#define DH 64

union FragBF {
    v16bf v;
    uint4 u4[2];
    u16   us[16];
};

__device__ __forceinline__ u16 f2bf(float f) {
    union { float f; unsigned u; } v;
    v.f = f;
    unsigned r = v.u + 0x7FFFu + ((v.u >> 16) & 1u);
    return (u16)(r >> 16);
}

__device__ __forceinline__ v8f wmma_bf16(const FragBF& a, const FragBF& b, v8f c) {
    return __builtin_amdgcn_wmma_f32_16x16x32_bf16(
        false, a.v, false, b.v, (short)0, c, false, false);
}

// async global -> LDS, 16 bytes per lane, tracked by ASYNCcnt
__device__ __forceinline__ void async_b128(void* lds_ptr, const u16* gptr) {
    unsigned lds_off = (unsigned)(unsigned long long)lds_ptr;  // low 32 bits = LDS byte offset
    unsigned long long gaddr = (unsigned long long)gptr;
    asm volatile("global_load_async_to_lds_b128 %0, %1, off"
                 :: "v"(lds_off), "v"(gaddr) : "memory");
}

// ---------------------------------------------------------------------------
// fp32 -> bf16 (raw bits) conversion
// ---------------------------------------------------------------------------
__global__ void __launch_bounds__(256) cvt_f32_bf16(const float* __restrict__ in,
                                                    u16* __restrict__ out, int n) {
    int i = blockIdx.x * 256 + threadIdx.x;
    if (i < n) out[i] = f2bf(in[i]);
}

// ---------------------------------------------------------------------------
// GEMM: C[m][n] = sum_k A[m][k] * Bt[n][k] + bias[n]
//   A  : bf16 [8192, 1024] row-major ; Bt : bf16 [1024, 1024] row-major
// mode 0: bf16 out as [B,H,S,DH] ; mode 1: bf16 out as [B,H,DH,S] (V^T)
// mode 2: f32 out [M,N] row-major.
// Block: 128 threads = 4 waves; block tile 128(M) x 64(N); wave tile 32 x 64.
// Register ping-pong double buffering over the K loop.
// ---------------------------------------------------------------------------
__global__ void __launch_bounds__(128)
gemm_bf16_wmma(const u16* __restrict__ A, const u16* __restrict__ Bt,
               const float* __restrict__ bias,
               u16* __restrict__ out_bf16, float* __restrict__ out_f32, int mode) {
    const int lane  = threadIdx.x & 31;
    const int w     = threadIdx.x >> 5;
    const int nlo   = lane & 15;
    const int khalf = lane >> 4;

    const int gm0 = blockIdx.x * 128 + w * 32;
    const int gn0 = blockIdx.y * 64;

    const u16* arow0 = A + (size_t)(gm0 + nlo) * DD;
    const u16* arow1 = A + (size_t)(gm0 + 16 + nlo) * DD;
    const u16* brow[4];
#pragma unroll
    for (int t = 0; t < 4; ++t) brow[t] = Bt + (size_t)(gn0 + t * 16 + nlo) * DD;

    v8f acc[2][4];
#pragma unroll
    for (int i = 0; i < 2; ++i)
#pragma unroll
        for (int t = 0; t < 4; ++t) acc[i][t] = (v8f){};

    auto LDA = [&](FragBF (&a)[2], int kk) {
        const int kb = kk + khalf * 8;
        a[0].u4[0] = *(const uint4*)(arow0 + kb);
        a[0].u4[1] = *(const uint4*)(arow0 + kb + 16);
        a[1].u4[0] = *(const uint4*)(arow1 + kb);
        a[1].u4[1] = *(const uint4*)(arow1 + kb + 16);
    };
    auto LDB = [&](FragBF (&b)[4], int kk) {
        const int kb = kk + khalf * 8;
#pragma unroll
        for (int t = 0; t < 4; ++t) {
            b[t].u4[0] = *(const uint4*)(brow[t] + kb);
            b[t].u4[1] = *(const uint4*)(brow[t] + kb + 16);
        }
    };
    auto MMA = [&](FragBF (&a)[2], FragBF (&b)[4]) {
#pragma unroll
        for (int t = 0; t < 4; ++t) {
            acc[0][t] = wmma_bf16(a[0], b[t], acc[0][t]);
            acc[1][t] = wmma_bf16(a[1], b[t], acc[1][t]);
        }
    };

    FragBF aP[2], bP[4], aQ[2], bQ[4];
    LDA(aP, 0);
    LDB(bP, 0);
    for (int kk = 0; kk < DD; kk += 64) {
        LDA(aQ, kk + 32);
        LDB(bQ, kk + 32);
        MMA(aP, bP);
        if (kk + 64 < DD) {
            LDA(aP, kk + 64);
            LDB(bP, kk + 64);
        }
        MMA(aQ, bQ);
    }

#pragma unroll
    for (int i = 0; i < 2; ++i) {
#pragma unroll
        for (int t = 0; t < 4; ++t) {
            const int n  = gn0 + t * 16 + nlo;
            const float bn = bias[n];
#pragma unroll
            for (int r = 0; r < 8; ++r) {
                const int m = gm0 + i * 16 + khalf * 8 + r;
                const float val = acc[i][t][r] + bn;
                if (mode == 2) {
                    out_f32[(size_t)m * DD + n] = val;
                } else {
                    const int b  = m >> 11;
                    const int s  = m & (SS - 1);
                    const int h  = n >> 6;
                    const int dh = n & (DH - 1);
                    size_t idx;
                    if (mode == 0)
                        idx = ((size_t)(b * HH + h) * SS + s) * DH + dh;   // [B,H,S,DH]
                    else
                        idx = ((size_t)(b * HH + h) * DH + dh) * SS + s;   // [B,H,DH,S]
                    out_bf16[idx] = f2bf(val);
                }
            }
        }
    }
}

// ---------------------------------------------------------------------------
// Flash attention. Block = 4 waves = 64 q rows of one (b,h).
// K/V 32-key tiles staged in LDS via async global->LDS, double buffered.
// ---------------------------------------------------------------------------
__global__ void __launch_bounds__(128)
attn_wmma(const u16* __restrict__ q, const u16* __restrict__ k,
          const u16* __restrict__ vT, u16* __restrict__ ctx) {
    __shared__ u16 sk[2][32 * 64];   // [buf][key row][d]        8 KB
    __shared__ u16 sv[2][64 * 32];   // [buf][d row][key]        8 KB
    __shared__ u16 sp[4][16 * 32];   // per-wave P tile          4 KB

    const int lane  = threadIdx.x & 31;
    const int w     = threadIdx.x >> 5;
    const int nlo   = lane & 15;
    const int khalf = lane >> 4;

    const int bh = blockIdx.y;
    const int q0 = blockIdx.x * 64 + w * 16;

    const u16* qh = q  + (size_t)bh * SS * DH;
    const u16* kh = k  + (size_t)bh * SS * DH;
    const u16* vh = vT + (size_t)bh * DH * SS;
    u16* plds = sp[w];

    // cooperative async stage of one 32-key tile (4 async loads per thread)
    auto stage = [&](int kk, int buf) {
        const int t = threadIdx.x;
#pragma unroll
        for (int c = t; c < 256; c += 128) {           // K tile: 32x64 bf16
            const int row = c >> 3, col = (c & 7) * 8;
            async_b128(&sk[buf][row * 64 + col], kh + (size_t)(kk + row) * DH + col);
        }
#pragma unroll
        for (int c = t; c < 256; c += 128) {           // V tile: 64x32 bf16
            const int row = c >> 2, col = (c & 3) * 8;
            async_b128(&sv[buf][row * 32 + col], vh + (size_t)row * SS + kk + col);
        }
    };

    // resident Q fragments: 16 rows x 64
    FragBF qa[2];
#pragma unroll
    for (int dblk = 0; dblk < 2; ++dblk) {
        const int row = q0 + nlo;
        const int db  = dblk * 32 + khalf * 8;
        qa[dblk].u4[0] = *(const uint4*)(qh + (size_t)row * DH + db);
        qa[dblk].u4[1] = *(const uint4*)(qh + (size_t)row * DH + db + 16);
    }

    v8f acc[4];
#pragma unroll
    for (int t = 0; t < 4; ++t) acc[t] = (v8f){};
    float mrow[8], lrow[8];
#pragma unroll
    for (int r = 0; r < 8; ++r) { mrow[r] = -3.0e38f; lrow[r] = 0.0f; }

    stage(0, 0);

    for (int kk = 0; kk < SS; kk += 32) {
        const int buf = (kk >> 5) & 1;
        if (kk + 32 < SS) {
            stage(kk + 32, buf ^ 1);
            asm volatile("s_wait_asynccnt 0x4" ::: "memory");
        } else {
            asm volatile("s_wait_asynccnt 0x0" ::: "memory");
        }
        __syncthreads();   // all waves' async portions of `buf` visible

        // ---- scores: two 16x16 tiles over 32 keys (B frags from LDS) ------
        v8f s0 = (v8f){}, s1 = (v8f){};
#pragma unroll
        for (int dblk = 0; dblk < 2; ++dblk) {
            const int kb = dblk * 32 + khalf * 8;
            FragBF b0, b1;
            b0.u4[0] = *(const uint4*)(&sk[buf][(nlo) * 64 + kb]);
            b0.u4[1] = *(const uint4*)(&sk[buf][(nlo) * 64 + kb + 16]);
            b1.u4[0] = *(const uint4*)(&sk[buf][(16 + nlo) * 64 + kb]);
            b1.u4[1] = *(const uint4*)(&sk[buf][(16 + nlo) * 64 + kb + 16]);
            s0 = wmma_bf16(qa[dblk], b0, s0);
            s1 = wmma_bf16(qa[dblk], b1, s1);
        }
        s0 *= 0.125f;   // 1/sqrt(64)
        s1 *= 0.125f;

        // ---- online softmax on D-fragment layout --------------------------
        float corr[8];
#pragma unroll
        for (int r = 0; r < 8; ++r) {
            float mx = fmaxf(s0[r], s1[r]);
#pragma unroll
            for (int off = 1; off < 16; off <<= 1)
                mx = fmaxf(mx, __shfl_xor(mx, off, 32));
            const float mnew = fmaxf(mrow[r], mx);
            corr[r] = __expf(mrow[r] - mnew);
            mrow[r] = mnew;
            const float p0 = __expf(s0[r] - mnew);
            const float p1 = __expf(s1[r] - mnew);
            s0[r] = p0;
            s1[r] = p1;
            float rs = p0 + p1;
#pragma unroll
            for (int off = 1; off < 16; off <<= 1)
                rs += __shfl_xor(rs, off, 32);
            lrow[r] = lrow[r] * corr[r] + rs;
        }
#pragma unroll
        for (int t = 0; t < 4; ++t)
#pragma unroll
            for (int r = 0; r < 8; ++r) acc[t][r] *= corr[r];

        // ---- transpose P: D-layout -> A-layout via per-wave LDS -----------
#pragma unroll
        for (int r = 0; r < 8; ++r) {
            const int row = khalf * 8 + r;
            plds[row * 32 + nlo]      = f2bf(s0[r]);
            plds[row * 32 + 16 + nlo] = f2bf(s1[r]);
        }
        __syncthreads();
        FragBF pa;
        pa.u4[0] = *(const uint4*)(plds + nlo * 32 + khalf * 8);
        pa.u4[1] = *(const uint4*)(plds + nlo * 32 + khalf * 8 + 16);

        // ---- ctx += P @ V (B frags from LDS V^T tile) ---------------------
#pragma unroll
        for (int t = 0; t < 4; ++t) {
            const int d  = t * 16 + nlo;
            const int kb = khalf * 8;
            FragBF bv;
            bv.u4[0] = *(const uint4*)(&sv[buf][d * 32 + kb]);
            bv.u4[1] = *(const uint4*)(&sv[buf][d * 32 + kb + 16]);
            acc[t] = wmma_bf16(pa, bv, acc[t]);
        }
        __syncthreads();   // protect `buf` before it is re-staged next iter
    }

    // ---- epilogue: normalize, write ctx as [B, S, D] bf16 -----------------
    const int b = bh >> 4;
    const int h = bh & 15;
#pragma unroll
    for (int t = 0; t < 4; ++t) {
#pragma unroll
        for (int r = 0; r < 8; ++r) {
            const int m = q0 + khalf * 8 + r;
            const float val = acc[t][r] / lrow[r];
            ctx[((size_t)(b * SS + m)) * DD + h * DH + t * 16 + nlo] = f2bf(val);
        }
    }
}

// ---------------------------------------------------------------------------
// Host-side orchestration
// ---------------------------------------------------------------------------
extern "C" void kernel_launch(void* const* d_in, const int* in_sizes, int n_in,
                              void* d_out, int out_size, void* d_ws, size_t ws_size,
                              hipStream_t stream) {
    const float* keys    = (const float*)d_in[0];
    const float* queries = (const float*)d_in[1];
    const float* values  = (const float*)d_in[2];
    const float* Wk = (const float*)d_in[3];
    const float* bk = (const float*)d_in[4];
    const float* Wq = (const float*)d_in[5];
    const float* bq = (const float*)d_in[6];
    const float* Wv = (const float*)d_in[7];
    const float* bv = (const float*)d_in[8];
    const float* Wo = (const float*)d_in[9];
    const float* bo = (const float*)d_in[10];
    float* out = (float*)d_out;

    const size_t MB = 1ull << 20;
    char* ws = (char*)d_ws;
    u16* xq  = (u16*)(ws + 0 * MB);
    u16* xk  = (u16*)(ws + 16 * MB);
    u16* xv  = (u16*)(ws + 32 * MB);
    u16* wq  = (u16*)(ws + 48 * MB);
    u16* wk  = (u16*)(ws + 50 * MB);
    u16* wv  = (u16*)(ws + 52 * MB);
    u16* wo  = (u16*)(ws + 54 * MB);
    u16* qh  = (u16*)(ws + 56 * MB);    // [B,H,S,DH]
    u16* kh  = (u16*)(ws + 72 * MB);    // [B,H,S,DH]
    u16* vT  = (u16*)(ws + 88 * MB);    // [B,H,DH,S]
    u16* ctx = (u16*)(ws + 104 * MB);   // [B,S,D]
    if (ws_size < 120 * MB) return;

    const int nX = BB * SS * DD;
    const int nW = DD * DD;

    cvt_f32_bf16<<<(nX + 255) / 256, 256, 0, stream>>>(queries, xq, nX);
    cvt_f32_bf16<<<(nX + 255) / 256, 256, 0, stream>>>(keys,    xk, nX);
    cvt_f32_bf16<<<(nX + 255) / 256, 256, 0, stream>>>(values,  xv, nX);
    cvt_f32_bf16<<<(nW + 255) / 256, 256, 0, stream>>>(Wq, wq, nW);
    cvt_f32_bf16<<<(nW + 255) / 256, 256, 0, stream>>>(Wk, wk, nW);
    cvt_f32_bf16<<<(nW + 255) / 256, 256, 0, stream>>>(Wv, wv, nW);
    cvt_f32_bf16<<<(nW + 255) / 256, 256, 0, stream>>>(Wo, wo, nW);

    dim3 gg((BB * SS) / 128, DD / 64);  // 64 x 16
    gemm_bf16_wmma<<<gg, 128, 0, stream>>>(xq, wq, bq, qh, nullptr, 0);
    gemm_bf16_wmma<<<gg, 128, 0, stream>>>(xk, wk, bk, kh, nullptr, 0);
    gemm_bf16_wmma<<<gg, 128, 0, stream>>>(xv, wv, bv, vT, nullptr, 1);

    attn_wmma<<<dim3(SS / 64, BB * HH), 128, 0, stream>>>(qh, kh, vT, ctx);

    gemm_bf16_wmma<<<gg, 128, 0, stream>>>(ctx, wo, bo, nullptr, out, 2);
}
